// Seq2SeqMBERT_44573170598498
// MI455X (gfx1250) — matrix operand order repaired and checked by reading, not compiled, not run on hardware
//
#include <hip/hip_runtime.h>

// ---------------------------------------------------------------------------
// Seq2Seq (BERT-emb + biLSTM encoder + Bahdanau attention decoder) on gfx1250.
// All matmuls: bf16 WMMA (v_wmma_f32_16x16x32_bf16, f32 accum).
// Bulk GEMMs use 1x4 N-tile register blocking (A-frag reused over 4 WMMAs).
// Recurrent scans: persistent 1024-thread (32-wave) single-WGP kernels with
// state in LDS; logits batched over all 63 steps as one [2016,256]x[256,32000].
// ---------------------------------------------------------------------------

typedef __bf16 bf16;
typedef __attribute__((ext_vector_type(16))) __bf16 v16bf;
typedef __attribute__((ext_vector_type(8)))  float  v8f;

#define Bsz   32
#define Ssz   128
#define Tsz   64
#define Vsz   32000
#define Dsz   768
#define Msz   256
#define Hsz   128

// ---------------- WMMA fragment helpers (16x16x32 bf16) ----------------
__device__ __forceinline__ v16bf load_a_frag(const bf16* __restrict__ base,
                                             int lda, int lane) {
  const int m = lane & 15, kh = (lane >> 4) & 1;
  const bf16* p = base + (size_t)m * lda + kh * 16;
  v16bf a;
#pragma unroll
  for (int i = 0; i < 8; ++i) a[i] = p[i];
#pragma unroll
  for (int i = 0; i < 8; ++i) a[i + 8] = p[i + 8];
  return a;
}

__device__ __forceinline__ v16bf load_b_frag(const bf16* __restrict__ base,
                                             int ldw, int lane) {
  const int n = lane & 15, kh = (lane >> 4) & 1;
  const bf16* p = base + (size_t)n * ldw + kh * 16;
  v16bf b;
#pragma unroll
  for (int i = 0; i < 16; ++i) b[i] = p[i];
  return b;
}

__device__ __forceinline__ v8f wmma_bf16(v16bf a, v16bf b, v8f c) {
  return __builtin_amdgcn_wmma_f32_16x16x32_bf16(
      false, a, false, b, (short)0, c, false, false);
}

__device__ __forceinline__ float sigf(float x) {
  return 1.0f / (1.0f + __expf(-x));
}

// ---------------- elementwise f32 -> bf16 ----------------
__global__ void f32_to_bf16_kernel(const float* __restrict__ src,
                                   bf16* __restrict__ dst, int n) {
  int i = blockIdx.x * 256 + threadIdx.x;
  if (i < n) dst[i] = (bf16)src[i];
}

// ---------------- embedding gather (time-major rows r = t*B + b) ----------
__global__ void gather_embed_kernel(const int* __restrict__ tok,
                                    const float* __restrict__ emb,
                                    bf16* __restrict__ out, int Bn, int Sn) {
  const int r = blockIdx.x;            // r = t*Bn + b
  const int t = r / Bn, b = r % Bn;
  const int token = tok[b * Sn + t];
  const float* src = emb + (size_t)token * Dsz;
  bf16* dst = out + (size_t)r * Dsz;
  for (int j = threadIdx.x; j < Dsz; j += 256) dst[j] = (bf16)src[j];
}

// ---------------- generic GEMM: C[M,N] = A[M,K] @ W[N,K]^T + bias ---------
// 8 waves/block; each wave computes FOUR adjacent 16x16 N-tiles, reusing the
// A fragment across 4 WMMAs per k-slice (N must be a multiple of 64).
// Optional row permutation on store: out_row = (row%permB)*permT + row/permB.
__global__ void __launch_bounds__(256)
gemm_bf16_kernel(const bf16* __restrict__ A, int lda,
                 const bf16* __restrict__ W, int ldw,
                 const float* __restrict__ bias,
                 float* __restrict__ Cf, bf16* __restrict__ Cb, int ldc,
                 int Mrows, int N, int K, int permB, int permT) {
  const int wave = threadIdx.x >> 5, lane = threadIdx.x & 31;
  const int groupsN = N >> 6;                       // groups of 4 N-tiles
  const long long groups = (long long)(Mrows >> 4) * groupsN;
  const long long id = (long long)blockIdx.x * 8 + wave;
  if (id >= groups) return;
  const int tm = (int)(id / groupsN);
  const int tn0 = (int)(id % groupsN) * 4;
  const int n = lane & 15, mh = lane >> 4;

  v8f acc[4];
#pragma unroll
  for (int j = 0; j < 4; ++j) {
    const float bv = bias ? bias[(tn0 + j) * 16 + n] : 0.0f;
#pragma unroll
    for (int r = 0; r < 8; ++r) acc[j][r] = bv;
  }

  const bf16* Abase = A + (size_t)(tm * 16) * lda;
  for (int k0 = 0; k0 < K; k0 += 32) {
    v16bf a = load_a_frag(Abase + k0, lda, lane);
#pragma unroll
    for (int j = 0; j < 4; ++j) {
      const bf16* Wb = W + (size_t)((tn0 + j) * 16) * ldw + k0;
      __builtin_prefetch(Wb + (size_t)n * ldw + 64, 0, 3);
      v16bf b = load_b_frag(Wb, ldw, lane);
      acc[j] = wmma_bf16(a, b, acc[j]);
    }
  }
#pragma unroll
  for (int j = 0; j < 4; ++j) {
#pragma unroll
    for (int r = 0; r < 8; ++r) {
      const int row = tm * 16 + r + 8 * mh;
      const long long orow =
          permB ? ((long long)(row % permB) * permT + row / permB) : row;
      const long long ci = orow * (long long)ldc + (tn0 + j) * 16 + n;
      const float v = acc[j][r];
      if (Cf) Cf[ci] = v;
      if (Cb) Cb[ci] = (bf16)v;
    }
  }
}

// ---------------- persistent encoder LSTM scan (one 32-wave WG per dir) ---
#define ENC_SMEM (32 * 512 * 4 + 32 * 128 * 4 + 32 * 128 * 4 + 32 * 128 * 2)
#define ENC_THREADS 1024
__global__ void __launch_bounds__(ENC_THREADS)
enc_scan_kernel(const float* __restrict__ xgF, const float* __restrict__ xgB,
                const bf16* __restrict__ WhhF, const bf16* __restrict__ WhhB,
                bf16* __restrict__ outB, float* __restrict__ outF,
                float* __restrict__ hFin, float* __restrict__ cFin, int S) {
  const int dir = blockIdx.x;                    // 0 = fw, 1 = bw
  const float* xg = dir ? xgB : xgF;
  const bf16* Whh = dir ? WhhB : WhhF;

  extern __shared__ char smE[];
  float* gates = (float*)smE;                    // [32,512]
  float* hF = gates + 32 * 512;                  // [32,128]
  float* cF = hF + 32 * Hsz;                     // [32,128]
  bf16* hB = (bf16*)(cF + 32 * Hsz);             // [32,128] bf16 (WMMA A)

  const int tid = threadIdx.x;
  const int wave = tid >> 5, lane = tid & 31;
  for (int e = tid; e < 32 * Hsz; e += ENC_THREADS) {
    hF[e] = 0.0f; cF[e] = 0.0f; hB[e] = (bf16)0.0f;
  }
  __syncthreads();

  for (int idx = 0; idx < S; ++idx) {
    const int t = dir ? (S - 1 - idx) : idx;
    // --- gates = xg[t] + h @ Whh^T  (2x32 tiles, K=128, 2 tiles/wave) ---
    for (int tile = wave; tile < 64; tile += 32) {
      const int tm = tile >> 5, tn = tile & 31;
      const int n = lane & 15, mh = lane >> 4;
      v8f acc;
      const float* xs = xg + (size_t)(t * Bsz + tm * 16) * 512 + tn * 16;
#pragma unroll
      for (int r = 0; r < 8; ++r) acc[r] = xs[(size_t)(r + 8 * mh) * 512 + n];
#pragma unroll
      for (int kk = 0; kk < 4; ++kk) {
        v16bf a = load_a_frag(hB + (tm * 16) * Hsz + kk * 32, Hsz, lane);
        v16bf b = load_b_frag(Whh + (size_t)(tn * 16) * Hsz + kk * 32, Hsz, lane);
        acc = wmma_bf16(a, b, acc);
      }
      float* g = gates + (tm * 16) * 512 + tn * 16;
#pragma unroll
      for (int r = 0; r < 8; ++r) g[(r + 8 * mh) * 512 + n] = acc[r];
    }
    __syncthreads();
    // --- fused LSTM cell, write layer output (concat column dir*128) ---
    for (int e = tid; e < 32 * Hsz; e += ENC_THREADS) {
      const int b = e >> 7, j = e & 127;
      const float gi = gates[b * 512 + j];
      const float gf = gates[b * 512 + Hsz + j];
      const float gg = gates[b * 512 + 2 * Hsz + j];
      const float go = gates[b * 512 + 3 * Hsz + j];
      const float cN = sigf(gf) * cF[e] + sigf(gi) * tanhf(gg);
      const float hN = sigf(go) * tanhf(cN);
      cF[e] = cN; hF[e] = hN; hB[e] = (bf16)hN;
      const size_t o = (size_t)(t * Bsz + b) * Msz + dir * Hsz + j;
      outB[o] = (bf16)hN;
      if (outF) outF[o] = hN;
    }
    __syncthreads();
  }
  for (int e = tid; e < 32 * Hsz; e += ENC_THREADS) {
    const int b = e >> 7, j = e & 127;
    hFin[b * Msz + dir * Hsz + j] = hF[e];
    cFin[b * Msz + dir * Hsz + j] = cF[e];
  }
}

// ---------------- persistent decoder scan (single 32-wave WG, 63 steps) ---
#define DEC_SMEM (32 * 1024 * 4 + 3 * 32 * 256 * 2 + 32 * 128 * 4 + 256 * 4)
#define DEC_THREADS 1024
__global__ void __launch_bounds__(DEC_THREADS)
dec_scan_kernel(const float* __restrict__ xgd,    // [2016,1024] (incl. b0)
                const bf16* __restrict__ Wih0c,   // dec0 Wih cols 256.., ldw=512
                const bf16* __restrict__ Whh0,    // [1024,256]
                const bf16* __restrict__ Wih1,    // [1024,256]
                const bf16* __restrict__ Whh1,    // [1024,256]
                const float* __restrict__ b1,     // [1024]
                const bf16* __restrict__ Wd,      // [256,256]
                const float* __restrict__ Wd_b,   // [256]
                const float* __restrict__ vw,     // [256]
                const float* __restrict__ vb,     // [1]
                const float* __restrict__ encKeys,// [4096,256] time-major
                const float* __restrict__ encOut, // [4096,256] f32
                const unsigned char* __restrict__ mask, // [32,128]
                const float* __restrict__ hI,     // [2,32,256]
                const float* __restrict__ cI,     // [2,32,256]
                float* __restrict__ cWS,          // [2,32,256] scratch
                bf16* __restrict__ hTop,          // [2016,256]
                int TT) {
  extern __shared__ char smD[];
  float* gates = (float*)smD;                      // [32,1024]; front = dec_q
  bf16* h0b = (bf16*)(gates + 32 * 1024);          // [32,256]
  bf16* h1b = h0b + 32 * Msz;                      // [32,256]
  bf16* ctxb = h1b + 32 * Msz;                     // [32,256]
  float* attn = (float*)(ctxb + 32 * Msz);         // [32,128]
  float* vv = attn + 32 * Ssz;                     // [256]

  const int tid = threadIdx.x;
  const int wave = tid >> 5, lane = tid & 31;
  for (int e = tid; e < 32 * Msz; e += DEC_THREADS) {
    h0b[e] = (bf16)hI[e];
    h1b[e] = (bf16)hI[32 * Msz + e];
    cWS[e] = cI[e];
    cWS[32 * Msz + e] = cI[32 * Msz + e];
  }
  if (tid < 256) vv[tid] = vw[tid];
  __syncthreads();

  const float vB = vb[0];
  const int n = lane & 15, mh = lane >> 4;
  for (int t = 0; t < TT; ++t) {
    // 1) dec_q = h1 @ Wd^T + Wd_b -> gates[0..32*256)  (one tile per wave)
    {
      const int tm = wave >> 4, tn = wave & 15;
      v8f acc;
      const float bvv = Wd_b[tn * 16 + n];
#pragma unroll
      for (int r = 0; r < 8; ++r) acc[r] = bvv;
#pragma unroll
      for (int kk = 0; kk < 8; ++kk) {
        v16bf a = load_a_frag(h1b + (tm * 16) * Msz + kk * 32, Msz, lane);
        v16bf b = load_b_frag(Wd + (size_t)(tn * 16) * Msz + kk * 32, Msz, lane);
        acc = wmma_bf16(a, b, acc);
      }
#pragma unroll
      for (int r = 0; r < 8; ++r)
        gates[(tm * 16 + r + 8 * mh) * Msz + tn * 16 + n] = acc[r];
    }
    __syncthreads();
    // 2) scores[b,s] = v . tanh(dec_q[b] + enc_keys[s,b]) + v_b
    for (int p = tid; p < 32 * Ssz; p += DEC_THREADS) {
      const int b = p >> 7, s = p & 127;
      const float* dq = gates + b * Msz;
      const float* ek = encKeys + (size_t)(s * Bsz + b) * Msz;
      float acc = vB;
      for (int k = 0; k < Msz; ++k) acc += vv[k] * tanhf(dq[k] + ek[k]);
      if (mask[b * Ssz + s]) acc = -1e10f;
      attn[p] = acc;
    }
    __syncthreads();
    // 3) softmax over s (one thread per batch row)
    if (tid < 32) {
      float* row = attn + tid * Ssz;
      float mx = row[0];
      for (int s = 1; s < Ssz; ++s) mx = fmaxf(mx, row[s]);
      float sm = 0.0f;
      for (int s = 0; s < Ssz; ++s) { float e = __expf(row[s] - mx); row[s] = e; sm += e; }
      const float inv = 1.0f / sm;
      for (int s = 0; s < Ssz; ++s) row[s] *= inv;
    }
    __syncthreads();
    // 4) context[b,m] = sum_s attn[b,s] * encOut[s,b,m]
    for (int p = tid; p < 32 * Msz; p += DEC_THREADS) {
      const int b = p >> 8, m = p & 255;
      const float* ar = attn + b * Ssz;
      float acc = 0.0f;
      for (int s = 0; s < Ssz; ++s)
        acc += ar[s] * encOut[(size_t)(s * Bsz + b) * Msz + m];
      ctxb[p] = (bf16)acc;
    }
    __syncthreads();
    // 5) gates0 = xgd[t] + ctx @ Wih0c^T + h0 @ Whh0^T
    //    32 groups of 4 adjacent N-tiles; one group per wave, A-frag reused.
    {
      const int tm = wave >> 4, tn0 = (wave & 15) * 4;
      v8f acc[4];
#pragma unroll
      for (int j = 0; j < 4; ++j) {
        const float* xs =
            xgd + (size_t)(t * Bsz + tm * 16) * 1024 + (tn0 + j) * 16;
#pragma unroll
        for (int r = 0; r < 8; ++r) acc[j][r] = xs[(size_t)(r + 8 * mh) * 1024 + n];
      }
#pragma unroll
      for (int kk = 0; kk < 8; ++kk) {
        v16bf a = load_a_frag(ctxb + (tm * 16) * Msz + kk * 32, Msz, lane);
#pragma unroll
        for (int j = 0; j < 4; ++j) {
          v16bf b = load_b_frag(Wih0c + (size_t)((tn0 + j) * 16) * 512 + kk * 32,
                                512, lane);
          acc[j] = wmma_bf16(a, b, acc[j]);
        }
      }
#pragma unroll
      for (int kk = 0; kk < 8; ++kk) {
        v16bf a = load_a_frag(h0b + (tm * 16) * Msz + kk * 32, Msz, lane);
#pragma unroll
        for (int j = 0; j < 4; ++j) {
          v16bf b = load_b_frag(Whh0 + (size_t)((tn0 + j) * 16) * Msz + kk * 32,
                                Msz, lane);
          acc[j] = wmma_bf16(a, b, acc[j]);
        }
      }
#pragma unroll
      for (int j = 0; j < 4; ++j)
#pragma unroll
        for (int r = 0; r < 8; ++r)
          gates[(tm * 16 + r + 8 * mh) * 1024 + (tn0 + j) * 16 + n] = acc[j][r];
    }
    __syncthreads();
    // 6) LSTM cell 0
    for (int e = tid; e < 32 * Msz; e += DEC_THREADS) {
      const int b = e >> 8, j = e & 255;
      const float gi = gates[b * 1024 + j];
      const float gf = gates[b * 1024 + Msz + j];
      const float gg = gates[b * 1024 + 2 * Msz + j];
      const float go = gates[b * 1024 + 3 * Msz + j];
      const float cN = sigf(gf) * cWS[e] + sigf(gi) * tanhf(gg);
      cWS[e] = cN;
      h0b[e] = (bf16)(sigf(go) * tanhf(cN));
    }
    __syncthreads();
    // 7) gates1 = h0 @ Wih1^T + h1 @ Whh1^T + b1  (same grouped layout)
    {
      const int tm = wave >> 4, tn0 = (wave & 15) * 4;
      v8f acc[4];
#pragma unroll
      for (int j = 0; j < 4; ++j) {
        const float bvv = b1[(tn0 + j) * 16 + n];
#pragma unroll
        for (int r = 0; r < 8; ++r) acc[j][r] = bvv;
      }
#pragma unroll
      for (int kk = 0; kk < 8; ++kk) {
        v16bf a = load_a_frag(h0b + (tm * 16) * Msz + kk * 32, Msz, lane);
#pragma unroll
        for (int j = 0; j < 4; ++j) {
          v16bf b = load_b_frag(Wih1 + (size_t)((tn0 + j) * 16) * Msz + kk * 32,
                                Msz, lane);
          acc[j] = wmma_bf16(a, b, acc[j]);
        }
      }
#pragma unroll
      for (int kk = 0; kk < 8; ++kk) {
        v16bf a = load_a_frag(h1b + (tm * 16) * Msz + kk * 32, Msz, lane);
#pragma unroll
        for (int j = 0; j < 4; ++j) {
          v16bf b = load_b_frag(Whh1 + (size_t)((tn0 + j) * 16) * Msz + kk * 32,
                                Msz, lane);
          acc[j] = wmma_bf16(a, b, acc[j]);
        }
      }
#pragma unroll
      for (int j = 0; j < 4; ++j)
#pragma unroll
        for (int r = 0; r < 8; ++r)
          gates[(tm * 16 + r + 8 * mh) * 1024 + (tn0 + j) * 16 + n] = acc[j][r];
    }
    __syncthreads();
    // 8) LSTM cell 1 -> h1, hTop[t]
    for (int e = tid; e < 32 * Msz; e += DEC_THREADS) {
      const int b = e >> 8, j = e & 255;
      const float gi = gates[b * 1024 + j];
      const float gf = gates[b * 1024 + Msz + j];
      const float gg = gates[b * 1024 + 2 * Msz + j];
      const float go = gates[b * 1024 + 3 * Msz + j];
      const float cN = sigf(gf) * cWS[32 * Msz + e] + sigf(gi) * tanhf(gg);
      cWS[32 * Msz + e] = cN;
      const float hN = sigf(go) * tanhf(cN);
      h1b[e] = (bf16)hN;
      hTop[(size_t)(t * Bsz) * Msz + e] = (bf16)hN;
    }
    __syncthreads();
  }
}

// ---------------------------------------------------------------------------
extern "C" void kernel_launch(void* const* d_in, const int* in_sizes, int n_in,
                              void* d_out, int out_size, void* d_ws,
                              size_t ws_size, hipStream_t stream) {
  (void)in_sizes; (void)n_in; (void)out_size; (void)ws_size;
  const int* s_tok = (const int*)d_in[0];
  const int* t_tok = (const int*)d_in[1];
  const unsigned char* mask = (const unsigned char*)d_in[2];
  const float* emb    = (const float*)d_in[3];
  const float* proj_w = (const float*)d_in[4];
  const float* proj_b = (const float*)d_in[5];
  const float* encWih[4] = {(const float*)d_in[6],  (const float*)d_in[9],
                            (const float*)d_in[12], (const float*)d_in[15]};
  const float* encWhh[4] = {(const float*)d_in[7],  (const float*)d_in[10],
                            (const float*)d_in[13], (const float*)d_in[16]};
  const float* encB[4]   = {(const float*)d_in[8],  (const float*)d_in[11],
                            (const float*)d_in[14], (const float*)d_in[17]};
  const float* Wd_w = (const float*)d_in[18]; const float* Wd_b = (const float*)d_in[19];
  const float* We_w = (const float*)d_in[20]; const float* We_b = (const float*)d_in[21];
  const float* v_w  = (const float*)d_in[22]; const float* v_b  = (const float*)d_in[23];
  const float* d0Wih = (const float*)d_in[24]; const float* d0Whh = (const float*)d_in[25];
  const float* d0b   = (const float*)d_in[26];
  const float* d1Wih = (const float*)d_in[27]; const float* d1Whh = (const float*)d_in[28];
  const float* d1b   = (const float*)d_in[29];
  const float* fc_w  = (const float*)d_in[30]; const float* fc_b = (const float*)d_in[31];

  char* base = (char*)d_ws;
  size_t off = 0;
  auto alloc = [&](size_t bytes) -> char* {
    char* p = base + off;
    off += (bytes + 255) & ~(size_t)255;
    return p;
  };
  bf16* wbProj = (bf16*)alloc((size_t)Msz * Dsz * 2);
  bf16* wbEncWih[4]; bf16* wbEncWhh[4];
  for (int i = 0; i < 4; ++i) {
    wbEncWih[i] = (bf16*)alloc((size_t)512 * Msz * 2);
    wbEncWhh[i] = (bf16*)alloc((size_t)512 * Hsz * 2);
  }
  bf16* wbWd = (bf16*)alloc((size_t)Msz * Msz * 2);
  bf16* wbWe = (bf16*)alloc((size_t)Msz * Msz * 2);
  bf16* wbD0Wih = (bf16*)alloc((size_t)1024 * 512 * 2);
  bf16* wbD0Whh = (bf16*)alloc((size_t)1024 * Msz * 2);
  bf16* wbD1Wih = (bf16*)alloc((size_t)1024 * Msz * 2);
  bf16* wbD1Whh = (bf16*)alloc((size_t)1024 * Msz * 2);
  bf16* wbFc = (bf16*)alloc((size_t)Vsz * Msz * 2);

  bf16* srcEmb = (bf16*)alloc((size_t)Bsz * Ssz * Dsz * 2);
  bf16* tgtEmb = (bf16*)alloc((size_t)Bsz * Tsz * Dsz * 2);
  bf16* srcB   = (bf16*)alloc((size_t)Bsz * Ssz * Msz * 2);
  bf16* tgtB   = (bf16*)alloc((size_t)Bsz * Tsz * Msz * 2);
  float* xgE[4];
  for (int i = 0; i < 4; ++i) xgE[i] = (float*)alloc((size_t)Bsz * Ssz * 512 * 4);
  bf16*  l0out   = (bf16*)alloc((size_t)Bsz * Ssz * Msz * 2);
  bf16*  encOutB = (bf16*)alloc((size_t)Bsz * Ssz * Msz * 2);
  float* encOutF = (float*)alloc((size_t)Bsz * Ssz * Msz * 4);
  float* encKeys = (float*)alloc((size_t)Bsz * Ssz * Msz * 4);
  float* xgd = (float*)alloc((size_t)(Tsz - 1) * Bsz * 1024 * 4);
  float* hI = (float*)alloc((size_t)2 * Bsz * Msz * 4);
  float* cI = (float*)alloc((size_t)2 * Bsz * Msz * 4);
  float* cWS = (float*)alloc((size_t)2 * Bsz * Msz * 4);
  bf16* hTop = (bf16*)alloc((size_t)(Tsz - 1) * Bsz * Msz * 2);

  auto conv = [&](const float* s, bf16* d, int n) {
    f32_to_bf16_kernel<<<(n + 255) / 256, 256, 0, stream>>>(s, d, n);
  };
  conv(proj_w, wbProj, Msz * Dsz);
  for (int i = 0; i < 4; ++i) {
    conv(encWih[i], wbEncWih[i], 512 * Msz);
    conv(encWhh[i], wbEncWhh[i], 512 * Hsz);
  }
  conv(Wd_w, wbWd, Msz * Msz);
  conv(We_w, wbWe, Msz * Msz);
  conv(d0Wih, wbD0Wih, 1024 * 512);
  conv(d0Whh, wbD0Whh, 1024 * Msz);
  conv(d1Wih, wbD1Wih, 1024 * Msz);
  conv(d1Whh, wbD1Whh, 1024 * Msz);
  conv(fc_w, wbFc, Vsz * Msz);

  gather_embed_kernel<<<Bsz * Ssz, 256, 0, stream>>>(s_tok, emb, srcEmb, Bsz, Ssz);
  gather_embed_kernel<<<Bsz * Tsz, 256, 0, stream>>>(t_tok, emb, tgtEmb, Bsz, Tsz);

  auto gemm = [&](const bf16* A, int lda, const bf16* W, int ldw,
                  const float* bias, float* Cf, bf16* Cb, int ldc, int Mr,
                  int N, int K, int pB, int pT) {
    long long groups = (long long)(Mr / 16) * (N / 64);
    int blocks = (int)((groups + 7) / 8);
    gemm_bf16_kernel<<<blocks, 256, 0, stream>>>(A, lda, W, ldw, bias, Cf, Cb,
                                                 ldc, Mr, N, K, pB, pT);
  };

  // projections
  gemm(srcEmb, Dsz, wbProj, Dsz, proj_b, nullptr, srcB, Msz, Bsz * Ssz, Msz, Dsz, 0, 0);
  gemm(tgtEmb, Dsz, wbProj, Dsz, proj_b, nullptr, tgtB, Msz, Bsz * Tsz, Msz, Dsz, 0, 0);
  // encoder layer 0
  gemm(srcB, Msz, wbEncWih[0], Msz, encB[0], xgE[0], nullptr, 512, Bsz * Ssz, 512, Msz, 0, 0);
  gemm(srcB, Msz, wbEncWih[1], Msz, encB[1], xgE[1], nullptr, 512, Bsz * Ssz, 512, Msz, 0, 0);
  hipFuncSetAttribute(reinterpret_cast<const void*>(enc_scan_kernel),
                      hipFuncAttributeMaxDynamicSharedMemorySize, ENC_SMEM);
  enc_scan_kernel<<<2, ENC_THREADS, ENC_SMEM, stream>>>(
      xgE[0], xgE[1], wbEncWhh[0], wbEncWhh[1], l0out, nullptr, hI, cI, Ssz);
  // encoder layer 1
  gemm(l0out, Msz, wbEncWih[2], Msz, encB[2], xgE[2], nullptr, 512, Bsz * Ssz, 512, Msz, 0, 0);
  gemm(l0out, Msz, wbEncWih[3], Msz, encB[3], xgE[3], nullptr, 512, Bsz * Ssz, 512, Msz, 0, 0);
  enc_scan_kernel<<<2, ENC_THREADS, ENC_SMEM, stream>>>(
      xgE[2], xgE[3], wbEncWhh[2], wbEncWhh[3], encOutB, encOutF,
      hI + Bsz * Msz, cI + Bsz * Msz, Ssz);
  // attention keys
  gemm(encOutB, Msz, wbWe, Msz, We_b, encKeys, nullptr, Msz, Bsz * Ssz, Msz, Msz, 0, 0);
  // decoder x-gates (steps 0..62, includes dec0 bias)
  gemm(tgtB, Msz, wbD0Wih, 512, d0b, xgd, nullptr, 1024, (Tsz - 1) * Bsz, 1024, Msz, 0, 0);
  // decoder scan
  hipFuncSetAttribute(reinterpret_cast<const void*>(dec_scan_kernel),
                      hipFuncAttributeMaxDynamicSharedMemorySize, DEC_SMEM);
  dec_scan_kernel<<<1, DEC_THREADS, DEC_SMEM, stream>>>(
      xgd, wbD0Wih + 256, wbD0Whh, wbD1Wih, wbD1Whh, d1b, wbWd, Wd_b, v_w, v_b,
      encKeys, encOutF, mask, hI, cI, cWS, hTop, Tsz - 1);
  // logits: rows permuted t*B+b -> b*63+t
  gemm(hTop, Msz, wbFc, Msz, fc_b, (float*)d_out, nullptr, Vsz,
       (Tsz - 1) * Bsz, Vsz, Msz, Bsz, Tsz - 1);
}
